// DPFA_29454885716026
// MI455X (gfx1250) — matrix operand order repaired and checked by reading, not compiled, not run on hardware
//
#include <hip/hip_runtime.h>
#include <hip/hip_bf16.h>
#include <math.h>

typedef __attribute__((ext_vector_type(16))) _Float16 v16h;
typedef __attribute__((ext_vector_type(8)))  _Float16 v8h;
typedef __attribute__((ext_vector_type(8)))  float    v8f;

union Frag16 { v16h v; v8h p[2]; _Float16 h[16]; };
union Frag8f { v8f  v; float f[8]; };

#define HH    64
#define WWID  64
#define HWSZ  4096
#define CIN   128
#define COUT  128
#define BATCH 8
#define GRP   4
#define CING  32
#define KTAP  9

// A-fragment (16-bit A 16x32 layout): two contiguous 8-half runs at
// row + kbase + half*8 and row + kbase + 16 + half*8 (both 16B aligned).
__device__ __forceinline__ void load_afrag(const _Float16* base, Frag16& a) {
  a.p[0] = *(const v8h*)(base);
  a.p[1] = *(const v8h*)(base + 16);
}

__device__ __forceinline__ float fast_sigmoid(float s) {
  return __fdividef(1.0f, 1.0f + __expf(-s));
}

// ---------------------------------------------------------------------------
// Position-attention map: pa[o][pix], constant over batch.
// ---------------------------------------------------------------------------
__global__ void pa_kernel(const float* __restrict__ w1, const float* __restrict__ b1,
                          const float* __restrict__ w2, const float* __restrict__ b2,
                          float* __restrict__ pa) {
  int pix = blockIdx.x * blockDim.x + threadIdx.x;
  if (pix >= HWSZ) return;
  int y = pix >> 6, x = pix & 63;
  float hid[16];
  #pragma unroll
  for (int j = 0; j < 16; ++j) {
    float s = b1[j];
    #pragma unroll
    for (int ky = 0; ky < 3; ++ky) {
      #pragma unroll
      for (int kx = 0; kx < 3; ++kx) {
        int yy = y + ky - 1, xx = x + kx - 1;
        if (yy >= 0 && yy < HH && xx >= 0 && xx < WWID) {
          float gx = -1.0f + 2.0f * (float)xx / 63.0f;
          float gy = -1.0f + 2.0f * (float)yy / 63.0f;
          s += gx * w1[j * 18 + ky * 3 + kx] + gy * w1[j * 18 + 9 + ky * 3 + kx];
        }
      }
    }
    hid[j] = fmaxf(s, 0.0f);
  }
  for (int o = 0; o < COUT; ++o) {
    float s = b2[o];
    #pragma unroll
    for (int j = 0; j < 16; ++j) s += w2[o * 16 + j] * hid[j];
    pa[(size_t)o * HWSZ + pix] = fast_sigmoid(s);
  }
}

// ---------------------------------------------------------------------------
// Weight staging into tap-major f16 GEMM layouts:
//   woff16[m][t*128 + c] = w_off[m][c][t]   (m 27 -> padded 32)
//   wdc16 [o][t*32  + i] = w_dc [o][i][t]
// ---------------------------------------------------------------------------
__global__ void wconv_kernel(const float* __restrict__ w_off, const float* __restrict__ w_dc,
                             _Float16* __restrict__ woff16, _Float16* __restrict__ wdc16) {
  int tid = blockIdx.x * blockDim.x + threadIdx.x;
  if (tid >= 32 * 1152) return;
  int m = tid / 1152, r = tid - m * 1152;
  int t = r >> 7, c = r & 127;
  woff16[tid] = (m < 27) ? (_Float16)w_off[((size_t)m * 128 + c) * 9 + t] : (_Float16)0.0f;
  if (tid < 128 * 288) {
    int o = tid / 288, rr = tid - o * 288;
    int tt = rr >> 5, i = rr & 31;
    wdc16[tid] = (_Float16)w_dc[((size_t)o * 32 + i) * 9 + tt];
  }
}

// ---------------------------------------------------------------------------
// Offset/mask conv: implicit GEMM via v_wmma_f32_16x16x32_f16.
// K tap-major (kd = t*128 + c); one wave handles a 32-pixel tile
// (two N-tiles sharing the A fragments): 4 WMMA per 32-wide K-chunk.
// ---------------------------------------------------------------------------
__global__ void __launch_bounds__(128)
offs_kernel(const float* __restrict__ x, const _Float16* __restrict__ woff16,
            const float* __restrict__ b_off,
            float* __restrict__ py, float* __restrict__ px, float* __restrict__ msk) {
  int wave = blockIdx.x * 4 + (threadIdx.x >> 5);   // 1024 waves
  int lane = threadIdx.x & 31;
  int half = lane >> 4, ncol = lane & 15;
  int b  = wave >> 7;
  int p0 = (wave & 127) << 5;                       // 32-pixel tile, one row
  int h  = p0 >> 6;
  int w0 = (p0 & 63) + ncol;                        // N-tile 0 column
  int w1 = w0 + 16;                                 // N-tile 1 column
  const float* xb = x + (size_t)b * CIN * HWSZ;

  v8f c00 = {}, c01 = {}, c10 = {}, c11 = {};
  for (int kk = 0; kk < 36; ++kk) {
    int kbase = kk * 32;
    int t  = kk >> 2;
    int dy = t / 3 - 1, dx = t - (t / 3) * 3 - 1;
    int yy  = h + dy;
    int xx0 = w0 + dx, xx1 = w1 + dx;
    bool vy = (unsigned)yy < (unsigned)HH;
    bool v0 = vy && ((unsigned)xx0 < (unsigned)WWID);
    bool v1 = vy && ((unsigned)xx1 < (unsigned)WWID);
    int yc  = min(max(yy, 0), HH - 1);
    int xc0 = min(max(xx0, 0), WWID - 1);
    int xc1 = min(max(xx1, 0), WWID - 1);
    int cch = ((kk & 3) << 5) + half * 8;
    const float* pB0 = xb + (size_t)cch * HWSZ + yc * WWID + xc0;
    const float* pB1 = xb + (size_t)cch * HWSZ + yc * WWID + xc1;

    Frag16 bf0, bf1, a0, a1;
    #pragma unroll
    for (int e = 0; e < 16; ++e) {
      float u0 = pB0[(size_t)(e + (e & 8)) * HWSZ];
      float u1 = pB1[(size_t)(e + (e & 8)) * HWSZ];
      bf0.h[e] = (_Float16)(v0 ? u0 : 0.0f);
      bf1.h[e] = (_Float16)(v1 ? u1 : 0.0f);
    }
    load_afrag(woff16 + (size_t)ncol        * 1152 + kbase + half * 8, a0);
    load_afrag(woff16 + (size_t)(ncol + 16) * 1152 + kbase + half * 8, a1);

    c00 = __builtin_amdgcn_wmma_f32_16x16x32_f16(false, a0.v, false, bf0.v, (short)0, c00, false, false);
    c10 = __builtin_amdgcn_wmma_f32_16x16x32_f16(false, a1.v, false, bf0.v, (short)0, c10, false, false);
    c01 = __builtin_amdgcn_wmma_f32_16x16x32_f16(false, a0.v, false, bf1.v, (short)0, c01, false, false);
    c11 = __builtin_amdgcn_wmma_f32_16x16x32_f16(false, a1.v, false, bf1.v, (short)0, c11, false, false);
  }

  Frag8f f[2][2];
  f[0][0].v = c00; f[0][1].v = c10; f[1][0].v = c01; f[1][1].v = c11;
  #pragma unroll
  for (int nt = 0; nt < 2; ++nt) {
    int pix = p0 + nt * 16 + ncol;
    int wcol = (nt == 0) ? w0 : w1;
    #pragma unroll
    for (int r = 0; r < 8; ++r) {
      int mrow = r + 8 * half;
      #pragma unroll
      for (int mt = 0; mt < 2; ++mt) {
        int ch = mrow + 16 * mt;
        if (ch >= 27) continue;
        float z  = f[nt][mt].f[r] + b_off[ch];
        float rl = fmaxf(z, 0.0f);                  // om = relu(conv+b)
        if (ch < 18) {
          int k = ch >> 1;
          if ((ch & 1) == 0)
            py[((size_t)b * 9 + k) * HWSZ + pix] = (float)h + (float)(k / 3 - 1) + rl;
          else
            px[((size_t)b * 9 + k) * HWSZ + pix] = (float)wcol + (float)(k % 3 - 1) + rl;
        } else {
          msk[((size_t)b * 9 + (ch - 18)) * HWSZ + pix] = fast_sigmoid(rl);
        }
      }
    }
  }
}

// ---------------------------------------------------------------------------
// Deformable grouped conv + position attention.
// One block per (b, 32-pixel tile); shared bilinear corner table (mask
// folded); wave g computes M=32, N=32, K=288 (tap-major) via 4 WMMA/chunk.
// ---------------------------------------------------------------------------
__global__ void __launch_bounds__(128)
dconv_kernel(const float* __restrict__ x, const _Float16* __restrict__ wdc16,
             const float* __restrict__ b_dc,
             const float* __restrict__ py, const float* __restrict__ px,
             const float* __restrict__ msk, const float* __restrict__ pa,
             float* __restrict__ out) {
  __shared__ int   sIdx[KTAP * 32 * 4];
  __shared__ float sWgt[KTAP * 32 * 4];

  int b  = blockIdx.x >> 7;
  int p0 = (blockIdx.x & 127) << 5;
  int tid = threadIdx.x;

  // Phase 1: bilinear corner table, 9 taps x 32 pixels, mask folded in.
  for (int item = tid; item < KTAP * 32; item += 128) {
    int t = item >> 5, j = item & 31;
    int pix = p0 + j;
    size_t o = ((size_t)b * 9 + t) * HWSZ + pix;
    float pyv = py[o], pxv = px[o], mv = msk[o];
    float fy = floorf(pyv), fx = floorf(pxv);
    int iy = (int)fy, ix = (int)fx;
    float ly = pyv - fy, lx = pxv - fx;
    float w00 = (1.0f - ly) * (1.0f - lx) * mv;
    float w01 = (1.0f - ly) * lx * mv;
    float w10 = ly * (1.0f - lx) * mv;
    float w11 = ly * lx * mv;
    bool yv0 = (iy >= 0) & (iy < HH);
    bool yv1 = (iy + 1 >= 0) & (iy + 1 < HH);
    bool xv0 = (ix >= 0) & (ix < WWID);
    bool xv1 = (ix + 1 >= 0) & (ix + 1 < WWID);
    int base = item * 4;
    sIdx[base + 0] = (yv0 && xv0) ? iy * WWID + ix           : 0;
    sIdx[base + 1] = (yv0 && xv1) ? iy * WWID + ix + 1       : 0;
    sIdx[base + 2] = (yv1 && xv0) ? (iy + 1) * WWID + ix     : 0;
    sIdx[base + 3] = (yv1 && xv1) ? (iy + 1) * WWID + ix + 1 : 0;
    sWgt[base + 0] = (yv0 && xv0) ? w00 : 0.0f;
    sWgt[base + 1] = (yv0 && xv1) ? w01 : 0.0f;
    sWgt[base + 2] = (yv1 && xv0) ? w10 : 0.0f;
    sWgt[base + 3] = (yv1 && xv1) ? w11 : 0.0f;
  }
  __syncthreads();

  // Phase 2: per-group WMMA GEMM (M=32, N=32, K=288 tap-major)
  int lane = tid & 31, half = lane >> 4, ncol = lane & 15;
  int g = tid >> 5;
  const float* xbg = x + ((size_t)b * CIN + (size_t)g * CING) * HWSZ;

  v8f c00 = {}, c01 = {}, c10 = {}, c11 = {};
  for (int kk = 0; kk < 9; ++kk) {
    int ita = (kk * 32 + ncol) * 4;        // N-tile 0 column = ncol
    int itb = (kk * 32 + 16 + ncol) * 4;   // N-tile 1 column = 16+ncol
    int   i0a = sIdx[ita + 0], i1a = sIdx[ita + 1], i2a = sIdx[ita + 2], i3a = sIdx[ita + 3];
    float g0a = sWgt[ita + 0], g1a = sWgt[ita + 1], g2a = sWgt[ita + 2], g3a = sWgt[ita + 3];
    int   i0b = sIdx[itb + 0], i1b = sIdx[itb + 1], i2b = sIdx[itb + 2], i3b = sIdx[itb + 3];
    float g0b = sWgt[itb + 0], g1b = sWgt[itb + 1], g2b = sWgt[itb + 2], g3b = sWgt[itb + 3];
    const float* pb = xbg + (size_t)(half * 8) * HWSZ;

    Frag16 bf0, bf1, a0, a1;
    #pragma unroll
    for (int e = 0; e < 16; ++e) {
      const float* xc = pb + (size_t)(e + (e & 8)) * HWSZ;
      float va = g0a * xc[i0a] + g1a * xc[i1a] + g2a * xc[i2a] + g3a * xc[i3a];
      float vb = g0b * xc[i0b] + g1b * xc[i1b] + g2b * xc[i2b] + g3b * xc[i3b];
      bf0.h[e] = (_Float16)va;
      bf1.h[e] = (_Float16)vb;
    }
    load_afrag(wdc16 + ((size_t)g * CING + ncol)      * 288 + kk * 32 + half * 8, a0);
    load_afrag(wdc16 + ((size_t)g * CING + ncol + 16) * 288 + kk * 32 + half * 8, a1);

    c00 = __builtin_amdgcn_wmma_f32_16x16x32_f16(false, a0.v, false, bf0.v, (short)0, c00, false, false);
    c10 = __builtin_amdgcn_wmma_f32_16x16x32_f16(false, a1.v, false, bf0.v, (short)0, c10, false, false);
    c01 = __builtin_amdgcn_wmma_f32_16x16x32_f16(false, a0.v, false, bf1.v, (short)0, c01, false, false);
    c11 = __builtin_amdgcn_wmma_f32_16x16x32_f16(false, a1.v, false, bf1.v, (short)0, c11, false, false);
  }

  Frag8f f[2][2];
  f[0][0].v = c00; f[0][1].v = c10; f[1][0].v = c01; f[1][1].v = c11;
  #pragma unroll
  for (int nt = 0; nt < 2; ++nt) {
    int pix = p0 + nt * 16 + ncol;
    #pragma unroll
    for (int r = 0; r < 8; ++r) {
      int o0 = g * CING + r + 8 * half;
      float v0 = (f[nt][0].f[r] + b_dc[o0]) * pa[(size_t)o0 * HWSZ + pix];
      out[((size_t)b * COUT + o0) * HWSZ + pix] = v0;
      int o1 = o0 + 16;
      float v1 = (f[nt][1].f[r] + b_dc[o1]) * pa[(size_t)o1 * HWSZ + pix];
      out[((size_t)b * COUT + o1) * HWSZ + pix] = v1;
    }
  }
}

// ---------------------------------------------------------------------------
extern "C" void kernel_launch(void* const* d_in, const int* in_sizes, int n_in,
                              void* d_out, int out_size, void* d_ws, size_t ws_size,
                              hipStream_t stream) {
  const float* x     = (const float*)d_in[0];
  const float* w_off = (const float*)d_in[1];
  const float* b_off = (const float*)d_in[2];
  const float* w_dc  = (const float*)d_in[3];
  const float* b_dc  = (const float*)d_in[4];
  const float* w_pa1 = (const float*)d_in[5];
  const float* b_pa1 = (const float*)d_in[6];
  const float* w_pa2 = (const float*)d_in[7];
  const float* b_pa2 = (const float*)d_in[8];
  float* out = (float*)d_out;

  char* ws = (char*)d_ws;
  float* pa   = (float*)ws;                 ws += (size_t)COUT * HWSZ * 4;       // 2 MB
  float* py   = (float*)ws;                 ws += (size_t)BATCH * 9 * HWSZ * 4;  // 1.125 MB
  float* px   = (float*)ws;                 ws += (size_t)BATCH * 9 * HWSZ * 4;
  float* msk  = (float*)ws;                 ws += (size_t)BATCH * 9 * HWSZ * 4;
  _Float16* woff16 = (_Float16*)ws;         ws += (size_t)32 * 1152 * 2;         // 72 KB
  _Float16* wdc16  = (_Float16*)ws;         ws += (size_t)128 * 288 * 2;         // 72 KB
  (void)ws_size; (void)in_sizes; (void)n_in; (void)out_size;

  pa_kernel<<<HWSZ / 256, 256, 0, stream>>>(w_pa1, b_pa1, w_pa2, b_pa2, pa);
  wconv_kernel<<<(32 * 1152 + 255) / 256, 256, 0, stream>>>(w_off, w_dc, woff16, wdc16);
  offs_kernel<<<256, 128, 0, stream>>>(x, woff16, b_off, py, px, msk);          // 1024 waves
  dconv_kernel<<<BATCH * 128, 128, 0, stream>>>(x, wdc16, b_dc, py, px, msk, pa, out);
}
